// draw_62251255988859
// MI455X (gfx1250) — compile-verified, matrix operand order checked by reading, not run on hardware
//
#include <hip/hip_runtime.h>

// ---------------------------------------------------------------- types
typedef __attribute__((ext_vector_type(16))) _Float16 v16h;
typedef __attribute__((ext_vector_type(8)))  _Float16 v8h;
typedef __attribute__((ext_vector_type(8)))  float    v8f;

#define IMG    64
#define NZ     100
#define NZP    128               // NZ padded to WMMA K granule
#define ENC_H  800
#define DEC_H  1600
#define IN_SZ  (IMG * IMG)       // 4096
#define IMG3   (3 * IN_SZ)       // 12288
#define D_ENC  (2 * IMG3 + DEC_H)// 26176  (divisible by 32)
#define BATCH  64
#define TSTEPS 10

__device__ __forceinline__ float sigf(float x) { return 1.0f / (1.0f + expf(-x)); }

// ------------------------------------------------- WMMA fragment loader
// 16-bit 16x32 fragment, row-major source with leading dim `ld` (halves).
// ISA layout (CDNA5 ISA): lanes 0-15 -> rows, regs hold K with the
// 0..7/16..23 (lane<16) and 8..15/24..31 (lane>=16) interleave
// => two 16-byte loads per lane at +hi*8 and +16+hi*8 halves.
__device__ __forceinline__ v16h load_frag(const _Float16* __restrict__ tile,
                                          int ld, int lane)
{
    const int r  = lane & 15;
    const int ho = (lane >> 4) << 3;            // 0 or 8 halves
    const _Float16* p = tile + (size_t)r * ld + ho;
    union { v16h v; v8h h[2]; } u;
    u.h[0] = *(const v8h*)(p);
    u.h[1] = *(const v8h*)(p + 16);
    return u.v;
}

__device__ __forceinline__ void load_set(const _Float16* __restrict__ A,
                                         const _Float16* __restrict__ wb,
                                         int K, int k, int lane,
                                         v16h& bf, v16h (&af)[4])
{
    bf = load_frag(wb + k, K, lane);
#pragma unroll
    for (int m = 0; m < 4; ++m)
        af[m] = load_frag(A + (size_t)(m << 4) * K + k, K, lane);
}

__device__ __forceinline__ void do_wmma(v8f (&acc)[4], const v16h (&af)[4], v16h bf)
{
#pragma unroll
    for (int m = 0; m < 4; ++m)
        acc[m] = __builtin_amdgcn_wmma_f32_16x16x32_f16(
            false, af[m], false, bf, (short)0, acc[m], false, false);
}

// ------------------------------------------------- split-K WMMA GEMM
// Cpart[gridDim.y][64][N] = A[64,K]_f16 @ W[N,K]^T  over K chunk
// grid = (N/64, S). Block = 128 thr (4 waves); wave w owns N tile
// blockIdx.x*64 + w*16 and iterates the 4 M tiles reusing its B fragment,
// so each W element is fetched exactly once across the whole GEMM.
// K loop: explicit ping-pong double buffer, manually unrolled x2 so there
// are no inter-iteration register copies; loads of step s+1 overlap the
// 4 WMMAs of step s.
__global__ void __launch_bounds__(128)
gemm_wmma(const _Float16* __restrict__ A, const _Float16* __restrict__ W,
          float* __restrict__ Cpart, int N, int K, int Kchunk)
{
    const int lane = threadIdx.x & 31;
    const int wave = threadIdx.x >> 5;
    const int ncol = (blockIdx.x * 4 + wave) << 4;   // N % 64 == 0 -> EXEC all-1
    const int kb   = blockIdx.y * Kchunk;
    const int ke   = (kb + Kchunk < K) ? (kb + Kchunk) : K;

    const _Float16* wb = W + (size_t)ncol * K;
    v8f acc[4] = {};

    v16h bf0, bf1, af0[4], af1[4];
    load_set(A, wb, K, kb, lane, bf0, af0);

    int k   = kb + 32;
    int rem = ((ke - kb) >> 5) - 1;   // k-steps remaining beyond buffer 0

    while (rem >= 2) {
        // prefetch this lane's weight row ~16 steps ahead (global_prefetch_b8)
        __builtin_prefetch((const void*)(wb + (size_t)(lane & 15) * K + k + 512), 0, 1);
        load_set(A, wb, K, k, lane, bf1, af1);
        do_wmma(acc, af0, bf0);
        load_set(A, wb, K, k + 32, lane, bf0, af0);
        do_wmma(acc, af1, bf1);
        k += 64;
        rem -= 2;
    }
    if (rem == 1) {
        load_set(A, wb, K, k, lane, bf1, af1);
        do_wmma(acc, af0, bf0);
        do_wmma(acc, af1, bf1);
    } else {
        do_wmma(acc, af0, bf0);
    }

    // f32 C/D layout: lanes 0-15: (M=r, N=lane); lanes 16-31: (M=8+r, N=lane-16)
    float* cp = Cpart + (size_t)blockIdx.y * BATCH * N;
    const int nc = lane & 15;
    const int mo = (lane >> 4) << 3;
#pragma unroll
    for (int m = 0; m < 4; ++m)
#pragma unroll
        for (int r = 0; r < 8; ++r)
            cp[(size_t)((m << 4) + mo + r) * N + (size_t)(ncol + nc)] = acc[m][r];
}

// deterministic split-K reduction: C (=|+=) sum_s part[s] (+ bias)
__global__ void reduce_partials(const float* __restrict__ part,
                                const float* __restrict__ bias,
                                float* __restrict__ C, int N, int S, int accumulate)
{
    int i = blockIdx.x * blockDim.x + threadIdx.x;
    if (i >= BATCH * N) return;
    float s = 0.0f;
    for (int j = 0; j < S; ++j) s += part[(size_t)j * BATCH * N + i];
    int n = i - (i / N) * N;
    if (bias) s += bias[n];
    if (accumulate) C[i] += s; else C[i] = s;
}

// ------------------------------------------------- helpers
__global__ void cvt_f16(const float* __restrict__ src, _Float16* __restrict__ dst,
                        int K, int Kp, int total)
{
    int i = blockIdx.x * blockDim.x + threadIdx.x;
    if (i >= total) return;
    int n = i / Kp, k = i - n * Kp;
    dst[i] = (k < K) ? (_Float16)src[(size_t)n * K + k] : (_Float16)0.0f;
}

__global__ void zero_b16(uint4* __restrict__ p, int n16)
{
    int i = blockIdx.x * blockDim.x + threadIdx.x;
    if (i >= n16) return;
    uint4 z; z.x = 0u; z.y = 0u; z.z = 0u; z.w = 0u;
    p[i] = z;
}

// enc_in = [x_flat | x - sigmoid(canvas) | h_dec]  -> f16 [64, 26176]
__global__ void build_enc_in(const float* __restrict__ x,
                             const float* __restrict__ canvas,
                             const float* __restrict__ h_dec,
                             _Float16* __restrict__ enc_in)
{
    int i = blockIdx.x * blockDim.x + threadIdx.x;
    if (i >= BATCH * D_ENC) return;
    int b = i / D_ENC, col = i - b * D_ENC;
    float v;
    if (col < IMG3) {
        v = x[(size_t)b * IMG3 + col];
    } else if (col < 2 * IMG3) {
        int c2 = col - IMG3;
        v = x[(size_t)b * IMG3 + c2] - sigf(canvas[(size_t)b * IMG3 + c2]);
    } else {
        v = h_dec[(size_t)b * DEC_H + (col - 2 * IMG3)];
    }
    enc_in[i] = (_Float16)v;
}

// LSTM pointwise (gates ordered i,f,g,o along N), also emits f16 copy of h
__global__ void lstm_cell(const float* __restrict__ g,
                          float* __restrict__ h, float* __restrict__ c,
                          _Float16* __restrict__ h16, int H)
{
    int i = blockIdx.x * blockDim.x + threadIdx.x;
    if (i >= BATCH * H) return;
    int b = i / H, j = i - b * H;
    const float* gr = g + (size_t)b * 4 * H;
    float gi = sigf(gr[j]);
    float gf = sigf(gr[H + j]);
    float gg = tanhf(gr[2 * H + j]);
    float go = sigf(gr[3 * H + j]);
    float cn = gf * c[i] + gi * gg;
    float hn = go * tanhf(cn);
    c[i] = cn; h[i] = hn; h16[i] = (_Float16)hn;
}

// mu/logvar FC (+relu), reparameterized z (padded to 128, pad pre-zeroed)
__global__ void fc_z(const float* __restrict__ h_mu, const float* __restrict__ h_lv,
                     const float* __restrict__ Wmu, const float* __restrict__ bmu,
                     const float* __restrict__ Wlv, const float* __restrict__ blv,
                     const float* __restrict__ eps,
                     float* __restrict__ out_mu, float* __restrict__ out_lv,
                     _Float16* __restrict__ z16)
{
    int i = blockIdx.x * blockDim.x + threadIdx.x;
    if (i >= BATCH * NZ) return;
    int b = i / NZ, n = i - b * NZ;
    const float* hm = h_mu + (size_t)b * ENC_H;
    const float* hl = h_lv + (size_t)b * ENC_H;
    const float* wm = Wmu + (size_t)n * ENC_H;
    const float* wl = Wlv + (size_t)n * ENC_H;
    float sm = bmu[n], sl = blv[n];
    for (int k = 0; k < ENC_H; ++k) { sm += hm[k] * wm[k]; sl += hl[k] * wl[k]; }
    sm = fmaxf(sm, 0.0f);
    sl = fmaxf(sl, 0.0f);
    out_mu[i] = sm;
    out_lv[i] = sl;
    float z = eps[i] * expf(0.5f * sl) + sm;
    z16[(size_t)b * NZP + n] = (_Float16)z;
}

__global__ void sigmoid_out(const float* __restrict__ canvas, float* __restrict__ out)
{
    int i = blockIdx.x * blockDim.x + threadIdx.x;
    if (i >= BATCH * IMG3) return;
    out[i] = sigf(canvas[i]);
}

// ---------------------------------------------------------------- launch
extern "C" void kernel_launch(void* const* d_in, const int* in_sizes, int n_in,
                              void* d_out, int out_size, void* d_ws, size_t ws_size,
                              hipStream_t stream)
{
    (void)in_sizes; (void)n_in; (void)out_size; (void)ws_size;

    const float* x       = (const float*)d_in[0];
    const float* noise   = (const float*)d_in[1];
    const float* W_ih_mu = (const float*)d_in[2];
    const float* W_hh_mu = (const float*)d_in[3];
    const float* b_mu    = (const float*)d_in[4];
    const float* W_ih_lv = (const float*)d_in[5];
    const float* W_hh_lv = (const float*)d_in[6];
    const float* b_lv    = (const float*)d_in[7];
    const float* W_mu_fc = (const float*)d_in[8];
    const float* b_mu_fc = (const float*)d_in[9];
    const float* W_lv_fc = (const float*)d_in[10];
    const float* b_lv_fc = (const float*)d_in[11];
    const float* W_ih_de = (const float*)d_in[12];
    const float* W_hh_de = (const float*)d_in[13];
    const float* b_dec   = (const float*)d_in[14];
    const float* W_write = (const float*)d_in[15];
    const float* b_write = (const float*)d_in[16];

    float* out_c  = (float*)d_out;                       // [64,3,64,64]
    float* out_mu = out_c + (size_t)BATCH * IMG3;        // [10,64,100]
    float* out_lv = out_mu + (size_t)TSTEPS * BATCH * NZ;

    // ---- workspace carving (256B aligned)
    char* ws = (char*)d_ws;
    size_t off = 0;
    auto carve = [&](size_t bytes) -> void* {
        void* p = ws + off;
        off += (bytes + 255) & ~(size_t)255;
        return p;
    };
    _Float16* w_ih_mu_h = (_Float16*)carve((size_t)4 * ENC_H * D_ENC * 2);
    _Float16* w_hh_mu_h = (_Float16*)carve((size_t)4 * ENC_H * ENC_H * 2);
    _Float16* w_ih_lv_h = (_Float16*)carve((size_t)4 * ENC_H * D_ENC * 2);
    _Float16* w_hh_lv_h = (_Float16*)carve((size_t)4 * ENC_H * ENC_H * 2);
    _Float16* w_ihde_h  = (_Float16*)carve((size_t)4 * DEC_H * NZP * 2);
    _Float16* w_hhde_h  = (_Float16*)carve((size_t)4 * DEC_H * DEC_H * 2);
    _Float16* w_wr_h    = (_Float16*)carve((size_t)IMG3 * DEC_H * 2);
    _Float16* enc_in_h  = (_Float16*)carve((size_t)BATCH * D_ENC * 2);
    float* gates_mu = (float*)carve((size_t)BATCH * 4 * ENC_H * 4);
    float* gates_lv = (float*)carve((size_t)BATCH * 4 * ENC_H * 4);
    float* gates_de = (float*)carve((size_t)BATCH * 4 * DEC_H * 4);
    // split-K partials: max over GEMMs of S*64*N floats (16*64*3200 = 3.2M)
    float* partial  = (float*)carve((size_t)16 * BATCH * 3200 * 4);

    size_t state_begin = off;
    float* canvas   = (float*)carve((size_t)BATCH * IMG3 * 4);
    float* h_mu     = (float*)carve((size_t)BATCH * ENC_H * 4);
    float* c_mu     = (float*)carve((size_t)BATCH * ENC_H * 4);
    float* h_lv     = (float*)carve((size_t)BATCH * ENC_H * 4);
    float* c_lv     = (float*)carve((size_t)BATCH * ENC_H * 4);
    float* h_dec    = (float*)carve((size_t)BATCH * DEC_H * 4);
    float* c_dec    = (float*)carve((size_t)BATCH * DEC_H * 4);
    _Float16* h_mu_h  = (_Float16*)carve((size_t)BATCH * ENC_H * 2);
    _Float16* h_lv_h  = (_Float16*)carve((size_t)BATCH * ENC_H * 2);
    _Float16* h_dec_h = (_Float16*)carve((size_t)BATCH * DEC_H * 2);
    _Float16* z_h     = (_Float16*)carve((size_t)BATCH * NZP * 2);
    size_t state_end = off;

    // ---- zero all recurrent state + canvas + z pad (every call: deterministic)
    {
        int n16 = (int)((state_end - state_begin) >> 4);
        zero_b16<<<(n16 + 255) / 256, 256, 0, stream>>>((uint4*)(ws + state_begin), n16);
    }

    // ---- f32 -> f16 weight conversion (halves the per-step HBM stream)
    auto cvt = [&](const float* s, _Float16* d, int K, int Kp, int total) {
        cvt_f16<<<(total + 255) / 256, 256, 0, stream>>>(s, d, K, Kp, total);
    };
    cvt(W_ih_mu, w_ih_mu_h, D_ENC, D_ENC, 4 * ENC_H * D_ENC);
    cvt(W_hh_mu, w_hh_mu_h, ENC_H, ENC_H, 4 * ENC_H * ENC_H);
    cvt(W_ih_lv, w_ih_lv_h, D_ENC, D_ENC, 4 * ENC_H * D_ENC);
    cvt(W_hh_lv, w_hh_lv_h, ENC_H, ENC_H, 4 * ENC_H * ENC_H);
    cvt(W_ih_de, w_ihde_h,  NZ,    NZP,   4 * DEC_H * NZP);
    cvt(W_hh_de, w_hhde_h,  DEC_H, DEC_H, 4 * DEC_H * DEC_H);
    cvt(W_write, w_wr_h,    DEC_H, DEC_H, IMG3 * DEC_H);

    // split-K GEMM + deterministic reduction
    auto gemm_split = [&](const _Float16* A, const _Float16* W, const float* bias,
                          float* C, int N, int K, int accumulate, int Starget) {
        int steps = K / 32;
        int Kc = (steps + Starget - 1) / Starget;      // k-steps per chunk
        int S  = (steps + Kc - 1) / Kc;                // actual #chunks (all non-empty)
        dim3 grid(N / 64, S);
        gemm_wmma<<<grid, 128, 0, stream>>>(A, W, partial, N, K, Kc * 32);
        int total = BATCH * N;
        reduce_partials<<<(total + 255) / 256, 256, 0, stream>>>(
            partial, bias, C, N, S, accumulate);
    };

    // ---- T sequential DRAW steps
    for (int t = 0; t < TSTEPS; ++t) {
        build_enc_in<<<(BATCH * D_ENC + 255) / 256, 256, 0, stream>>>(
            x, canvas, h_dec, enc_in_h);

        // encoder mu / lv LSTM gate GEMMs
        gemm_split(enc_in_h, w_ih_mu_h, b_mu,    gates_mu, 4 * ENC_H, D_ENC, 0, 16);
        gemm_split(h_mu_h,   w_hh_mu_h, nullptr, gates_mu, 4 * ENC_H, ENC_H, 1, 4);
        gemm_split(enc_in_h, w_ih_lv_h, b_lv,    gates_lv, 4 * ENC_H, D_ENC, 0, 16);
        gemm_split(h_lv_h,   w_hh_lv_h, nullptr, gates_lv, 4 * ENC_H, ENC_H, 1, 4);

        lstm_cell<<<(BATCH * ENC_H + 255) / 256, 256, 0, stream>>>(
            gates_mu, h_mu, c_mu, h_mu_h, ENC_H);
        lstm_cell<<<(BATCH * ENC_H + 255) / 256, 256, 0, stream>>>(
            gates_lv, h_lv, c_lv, h_lv_h, ENC_H);

        fc_z<<<(BATCH * NZ + 255) / 256, 256, 0, stream>>>(
            h_mu, h_lv, W_mu_fc, b_mu_fc, W_lv_fc, b_lv_fc,
            noise + (size_t)t * BATCH * NZ,
            out_mu + (size_t)t * BATCH * NZ,
            out_lv + (size_t)t * BATCH * NZ, z_h);

        // decoder LSTM gate GEMMs
        gemm_split(z_h,     w_ihde_h, b_dec,   gates_de, 4 * DEC_H, NZP,   0, 1);
        gemm_split(h_dec_h, w_hhde_h, nullptr, gates_de, 4 * DEC_H, DEC_H, 1, 8);

        lstm_cell<<<(BATCH * DEC_H + 255) / 256, 256, 0, stream>>>(
            gates_de, h_dec, c_dec, h_dec_h, DEC_H);

        // canvas += h_dec @ W_write^T + b_write
        gemm_split(h_dec_h, w_wr_h, b_write, canvas, IMG3, DEC_H, 1, 4);
    }

    sigmoid_out<<<(BATCH * IMG3 + 255) / 256, 256, 0, stream>>>(canvas, out_c);
}
